// AdditiveAttentionSACModel_34179349741799
// MI455X (gfx1250) — matrix operand order repaired and checked by reading, not compiled, not run on hardware
//
#include <hip/hip_runtime.h>
#include <hip/hip_bf16.h>
#include <math.h>
#include <stdint.h>

// ---------------- model constants ----------------
#define B_TOTAL 32768
#define N_INT   32
#define OWN_D   3
#define INT_D   7
#define ATTN    128
#define HID     256
#define OUT_D   2
#define OBS_D   227          // OWN_D + N_INT*INT_D
#define SLOPE   0.2f

typedef __attribute__((ext_vector_type(16))) _Float16 v16h;
typedef __attribute__((ext_vector_type(8)))  _Float16 v8h;
typedef __attribute__((ext_vector_type(8)))  float    v8f;

// ---------------- workspace layout (in f16 elements) ----------------
#define WS_WQT   0                     // 128*128
#define WS_WKT   16384
#define WS_WVT   32768
#define WS_PROJT 49152
#define WS_H1T   65536                 // 256*256
#define WS_H2T   131072
#define WS_X     196608                // B*256  (concat[own_e, attn_vec]) f16

// ---------------- WMMA fragment helpers ----------------
__device__ __forceinline__ v16h frag_cat(v8h a, v8h b) {
    return __builtin_shufflevector(a, b, 0,1,2,3,4,5,6,7,8,9,10,11,12,13,14,15);
}

// A fragment: 16x32 (MxK) f16, row-major source with row stride `ld` halfs.
// Lane (m = lane&15, h = lane>>4): two contiguous 8-half chunks [h*8..+7], [16+h*8..+7].
__device__ __forceinline__ v16h load_fragA(const _Float16* base, int ld, int k0, int lane) {
    int m = lane & 15, h = lane >> 4;
    const _Float16* row = base + m * ld + k0;
    v8h a0 = *(const v8h*)(row + h * 8);
    v8h a1 = *(const v8h*)(row + 16 + h * 8);
    return frag_cat(a0, a1);
}

// B fragment: 32x16 (KxN) f16 from TRANSPOSED weight WT[N][K] (ld = K).
// Lane (n = lane&15, h = lane>>4): two contiguous 8-half chunks [h*8..+7], [16+h*8..+7].
__device__ __forceinline__ v16h load_fragB(const _Float16* baseT, int ld, int n0, int k0, int lane) {
    int n = lane & 15, h = lane >> 4;
    const _Float16* p = baseT + (size_t)(n0 + n) * ld + k0;
    return frag_cat(*(const v8h*)(p + h * 8), *(const v8h*)(p + 16 + h * 8));
}

__device__ __forceinline__ v8f wmma_f16(v16h a, v16h b, v8f c) {
    return __builtin_amdgcn_wmma_f32_16x16x32_f16(false, a, false, b, (short)0, c, false, false);
}

__device__ __forceinline__ float lrelu(float x) { return x > 0.f ? x : SLOPE * x; }

// Async global->LDS copy (GVS mode), tracked by ASYNCcnt.
__device__ __forceinline__ void async_load_b128(uint32_t lds_byte_addr, uint32_t voff_bytes,
                                                const void* sbase) {
    asm volatile("global_load_async_to_lds_b128 %0, %1, %2"
                 :: "v"(lds_byte_addr), "v"(voff_bytes), "s"(sbase)
                 : "memory");
}
__device__ __forceinline__ void wait_async0() {
    asm volatile("s_wait_asynccnt 0x0" ::: "memory");
}

// ---------------- kernel 0: fp32 -> f16 transpose (W[K][N] -> WT[N][K]) ----------------
__global__ void wconv_t_kernel(const float* __restrict__ W, _Float16* __restrict__ WT, int K, int N) {
    int i = blockIdx.x * blockDim.x + threadIdx.x;
    if (i < K * N) {
        int k = i / N, n = i % N;
        WT[(size_t)n * K + k] = (_Float16)W[i];
    }
}

// ---------------- kernel 1: embed + additive attention (16 batch rows / WG) ----------------
__global__ __launch_bounds__(256) void attn_embed_kernel(
    const float* __restrict__ obs,
    const float* __restrict__ own_W, const float* __restrict__ own_b,
    const float* __restrict__ int_W, const float* __restrict__ int_b,
    const float* __restrict__ v_att, const float* __restrict__ proj_b,
    const _Float16* __restrict__ WqT, const _Float16* __restrict__ WkT,
    const _Float16* __restrict__ WvT, const _Float16* __restrict__ projT,
    _Float16* __restrict__ x)
{
    extern __shared__ char smem[];
    float*    sObs   = (float*)smem;              // [16][232]
    float*    sIW    = sObs   + 16 * 232;         // [7][128]
    float*    sOW    = sIW    + 7 * 128;          // [3][128]
    float*    sIB    = sOW    + 3 * 128;          // [128]
    float*    sOB    = sIB    + 128;              // [128]
    float*    sVA    = sOB    + 128;              // [128]
    float*    sQ     = sVA    + 128;              // [16][132]
    float*    sScore = sQ     + 16 * 132;         // [16][32]
    float*    sAlpha = sScore + 16 * 32;          // [16][32]
    float*    sCtx   = sAlpha + 16 * 32;          // [16][132]
    _Float16* sOwnEh = (_Float16*)(sCtx + 16 * 132); // [16][136]
    _Float16* sCtxh  = sOwnEh + 16 * 136;         // [16][136]
    _Float16* sIntE  = sCtxh  + 16 * 136;         // [512][136]   (rows = b*32 + n)

    const int tid  = threadIdx.x;
    const int lane = tid & 31;
    const int wave = tid >> 5;
    const int b0   = blockIdx.x * 16;

    // ---- stage obs tile + small fp32 weights into LDS ----
    for (int i = tid; i < 16 * OBS_D; i += 256) {
        int r = i / OBS_D, c = i % OBS_D;
        sObs[r * 232 + c] = obs[(size_t)(b0 + r) * OBS_D + c];
    }
    for (int i = tid; i < 7 * 128; i += 256) sIW[i] = int_W[i];
    for (int i = tid; i < 3 * 128; i += 256) sOW[i] = own_W[i];
    if (tid < 128) { sIB[tid] = int_b[tid]; sOB[tid] = own_b[tid]; sVA[tid] = v_att[tid]; }
    for (int i = tid; i < 16 * 32;  i += 256) sScore[i] = 0.f;
    for (int i = tid; i < 16 * 132; i += 256) sCtx[i]   = 0.f;
    __syncthreads();

    // ---- own_e = lrelu(own @ own_W + b)  (K=3: VALU) ----
    for (int i = tid; i < 16 * 128; i += 256) {
        int r = i >> 7, d = i & 127;
        float acc = sOB[d];
        #pragma unroll
        for (int j = 0; j < OWN_D; ++j) acc += sObs[r * 232 + j] * sOW[j * 128 + d];
        acc = lrelu(acc);
        sOwnEh[r * 136 + d] = (_Float16)acc;
        x[(size_t)(b0 + r) * 256 + d] = (_Float16)acc;   // first half of concat
    }
    // ---- int_e = lrelu(intr @ int_W + b)  (K=7: VALU), f16 into WMMA-A layout ----
    for (int i = tid; i < 512 * 128; i += 256) {
        int row = i >> 7, d = i & 127;
        int r = row >> 5, n = row & 31;
        float acc = sIB[d];
        const float* ip = &sObs[r * 232 + OWN_D + n * INT_D];
        #pragma unroll
        for (int j = 0; j < INT_D; ++j) acc += ip[j] * sIW[j * 128 + d];
        sIntE[row * 136 + d] = (_Float16)lrelu(acc);
    }
    __syncthreads();

    // ---- q = own_e @ Wq : 1 row-tile x 8 col-tiles, one col-tile per wave ----
    {
        const int n0 = wave * 16;
        v8f c = {};
        #pragma unroll
        for (int kk = 0; kk < 4; ++kk) {
            v16h a = load_fragA(sOwnEh, 136, kk * 32, lane);
            v16h b = load_fragB(WqT, 128, n0, kk * 32, lane);
            c = wmma_f16(a, b, c);
        }
        int n = lane & 15, h = lane >> 4;
        #pragma unroll
        for (int r = 0; r < 8; ++r) sQ[(h * 8 + r) * 132 + n0 + n] = c[r];
    }
    __syncthreads();

    // ---- k tiles (never stored): each wave owns col-tile d0=wave*16, B frags hoisted ----
    {
        const int d0 = wave * 16;
        const int h = lane >> 4, d = d0 + (lane & 15);
        const float va = sVA[d];
        v16h bk[4];
        #pragma unroll
        for (int kk = 0; kk < 4; ++kk) bk[kk] = load_fragB(WkT, 128, d0, kk * 32, lane);
        for (int rt = 0; rt < 32; ++rt) {
            const int r0 = rt * 16;
            v8f c = {};
            #pragma unroll
            for (int kk = 0; kk < 4; ++kk)
                c = wmma_f16(load_fragA(sIntE + r0 * 136, 136, kk * 32, lane), bk[kk], c);
            #pragma unroll
            for (int r = 0; r < 8; ++r) {
                int m  = r0 + h * 8 + r;                 // row = b*32 + n
                int bb = m >> 5, n = m & 31;
                float e = tanhf(sQ[bb * 132 + d] + c[r]);
                atomicAdd(&sScore[bb * 32 + n], va * e);
            }
        }
    }
    __syncthreads();

    // ---- padding mask + softmax (+ nan_to_num) ----
    if (tid < 16 * 32) {
        int bb = tid >> 5, n = tid & 31;
        float s = 0.f;
        const float* ip = &sObs[bb * 232 + OWN_D + n * INT_D];
        #pragma unroll
        for (int j = 0; j < INT_D; ++j) s += fabsf(ip[j]);
        if (s < 1e-6f) sScore[tid] = -INFINITY;
    }
    __syncthreads();
    if (tid < 16) {
        float mx = -INFINITY;
        for (int n = 0; n < 32; ++n) mx = fmaxf(mx, sScore[tid * 32 + n]);
        if (mx > -INFINITY) {
            float sum = 0.f;
            for (int n = 0; n < 32; ++n) { float e = __expf(sScore[tid * 32 + n] - mx); sAlpha[tid * 32 + n] = e; sum += e; }
            float inv = 1.f / sum;
            for (int n = 0; n < 32; ++n) sAlpha[tid * 32 + n] *= inv;
        } else {
            for (int n = 0; n < 32; ++n) sAlpha[tid * 32 + n] = 0.f;   // all-padded row
        }
    }
    __syncthreads();

    // ---- v tiles (never stored): ctx[b,d] += alpha[b,n]*v[b,n,d], B frags hoisted ----
    {
        const int d0 = wave * 16;
        const int h = lane >> 4, d = d0 + (lane & 15);
        v16h bv[4];
        #pragma unroll
        for (int kk = 0; kk < 4; ++kk) bv[kk] = load_fragB(WvT, 128, d0, kk * 32, lane);
        for (int rt = 0; rt < 32; ++rt) {
            const int r0 = rt * 16;
            v8f c = {};
            #pragma unroll
            for (int kk = 0; kk < 4; ++kk)
                c = wmma_f16(load_fragA(sIntE + r0 * 136, 136, kk * 32, lane), bv[kk], c);
            #pragma unroll
            for (int r = 0; r < 8; ++r) {
                int m  = r0 + h * 8 + r;
                int bb = m >> 5, n = m & 31;
                atomicAdd(&sCtx[bb * 132 + d], sAlpha[bb * 32 + n] * c[r]);
            }
        }
    }
    __syncthreads();
    for (int i = tid; i < 16 * 128; i += 256) {
        int r = i >> 7, d = i & 127;
        sCtxh[r * 136 + d] = (_Float16)sCtx[r * 132 + d];
    }
    __syncthreads();

    // ---- attn_vec = tanh(ctx @ proj_W + proj_b), write second half of concat ----
    {
        const int d0 = wave * 16;
        v8f c = {};
        #pragma unroll
        for (int kk = 0; kk < 4; ++kk) {
            v16h a = load_fragA(sCtxh, 136, kk * 32, lane);
            v16h b = load_fragB(projT, 128, d0, kk * 32, lane);
            c = wmma_f16(a, b, c);
        }
        int h = lane >> 4, d = d0 + (lane & 15);
        float pb = proj_b[d];
        #pragma unroll
        for (int r = 0; r < 8; ++r) {
            float v = tanhf(c[r] + pb);
            x[(size_t)(b0 + h * 8 + r) * 256 + 128 + d] = (_Float16)v;
        }
    }
}

// ---------------- kernel 2: MLP head (64 batch rows / WG) ----------------
__global__ __launch_bounds__(256) void mlp_kernel(
    const _Float16* __restrict__ x,
    const _Float16* __restrict__ h1T, const _Float16* __restrict__ h2T,
    const float* __restrict__ h1_b, const float* __restrict__ h2_b,
    const float* __restrict__ out_W, const float* __restrict__ out_b,
    float* __restrict__ out)
{
    extern __shared__ char smem[];
    _Float16* sX = (_Float16*)smem;        // [64][264]
    _Float16* sH = sX + 64 * 264;          // [64][264]

    const int tid  = threadIdx.x;
    const int lane = tid & 31;
    const int wave = tid >> 5;
    const int b0   = blockIdx.x * 64;

    __builtin_prefetch(h1T, 0, 0);
    __builtin_prefetch(h2T, 0, 0);

    // stage x tile: async global->LDS b128 copies (ASYNCcnt path)
    {
        const void* gbase = (const void*)(x + (size_t)b0 * 256);
        for (int i = tid; i < 64 * 32; i += 256) {
            int r = i >> 5, c = (i & 31) * 8;
            uint32_t lds_addr = (uint32_t)(uintptr_t)(sX + r * 264 + c);
            uint32_t voff     = (uint32_t)((r * 256 + c) * sizeof(_Float16));
            async_load_b128(lds_addr, voff, gbase);
        }
        wait_async0();
    }
    __syncthreads();

    // h1 = lrelu(x @ h1_W + h1_b): each wave owns 2 col-tiles, B frags hoisted, K=256
    #pragma unroll
    for (int cc = 0; cc < 2; ++cc) {
        const int d0 = (wave * 2 + cc) * 16;
        const int h = lane >> 4, d = d0 + (lane & 15);
        v16h bw[8];
        #pragma unroll
        for (int kk = 0; kk < 8; ++kk) bw[kk] = load_fragB(h1T, 256, d0, kk * 32, lane);
        const float bias = h1_b[d];
        for (int rt = 0; rt < 4; ++rt) {
            v8f c = {};
            #pragma unroll
            for (int kk = 0; kk < 8; ++kk)
                c = wmma_f16(load_fragA(sX + rt * 16 * 264, 264, kk * 32, lane), bw[kk], c);
            #pragma unroll
            for (int r = 0; r < 8; ++r)
                sH[(rt * 16 + h * 8 + r) * 264 + d] = (_Float16)lrelu(c[r] + bias);
        }
    }
    __syncthreads();

    // h2 = lrelu(h1 @ h2_W + h2_b) -> overwrite sX
    #pragma unroll
    for (int cc = 0; cc < 2; ++cc) {
        const int d0 = (wave * 2 + cc) * 16;
        const int h = lane >> 4, d = d0 + (lane & 15);
        v16h bw[8];
        #pragma unroll
        for (int kk = 0; kk < 8; ++kk) bw[kk] = load_fragB(h2T, 256, d0, kk * 32, lane);
        const float bias = h2_b[d];
        for (int rt = 0; rt < 4; ++rt) {
            v8f c = {};
            #pragma unroll
            for (int kk = 0; kk < 8; ++kk)
                c = wmma_f16(load_fragA(sH + rt * 16 * 264, 264, kk * 32, lane), bw[kk], c);
            #pragma unroll
            for (int r = 0; r < 8; ++r)
                sX[(rt * 16 + h * 8 + r) * 264 + d] = (_Float16)lrelu(c[r] + bias);
        }
    }
    __syncthreads();

    // out = h2 @ out_W + out_b  (N=2: VALU)
    if (tid < 64 * OUT_D) {
        int row = tid >> 1, o = tid & 1;
        float acc = out_b[o];
        for (int d = 0; d < 256; ++d) acc += (float)sX[row * 264 + d] * out_W[d * OUT_D + o];
        out[(size_t)(b0 + row) * OUT_D + o] = acc;
    }
}

// ---------------- host launcher ----------------
extern "C" void kernel_launch(void* const* d_in, const int* in_sizes, int n_in,
                              void* d_out, int out_size, void* d_ws, size_t ws_size,
                              hipStream_t stream) {
    (void)in_sizes; (void)n_in; (void)out_size; (void)ws_size;

    const float* obs    = (const float*)d_in[0];
    const float* own_W  = (const float*)d_in[1];
    const float* own_b  = (const float*)d_in[2];
    const float* int_W  = (const float*)d_in[3];
    const float* int_b  = (const float*)d_in[4];
    const float* Wq     = (const float*)d_in[5];
    const float* Wk     = (const float*)d_in[6];
    const float* Wv     = (const float*)d_in[7];
    const float* v_att  = (const float*)d_in[8];
    const float* proj_W = (const float*)d_in[9];
    const float* proj_b = (const float*)d_in[10];
    const float* h1_W   = (const float*)d_in[11];
    const float* h1_b   = (const float*)d_in[12];
    const float* h2_W   = (const float*)d_in[13];
    const float* h2_b   = (const float*)d_in[14];
    const float* out_W  = (const float*)d_in[15];
    const float* out_b  = (const float*)d_in[16];
    float* out = (float*)d_out;

    _Float16* ws    = (_Float16*)d_ws;
    _Float16* WqT   = ws + WS_WQT;
    _Float16* WkT   = ws + WS_WKT;
    _Float16* WvT   = ws + WS_WVT;
    _Float16* projT = ws + WS_PROJT;
    _Float16* h1T   = ws + WS_H1T;
    _Float16* h2T   = ws + WS_H2T;
    _Float16* x     = ws + WS_X;

    // weight convert + transpose (fp32 -> f16 [N][K])
    wconv_t_kernel<<<(128 * 128 + 255) / 256, 256, 0, stream>>>(Wq, WqT, 128, 128);
    wconv_t_kernel<<<(128 * 128 + 255) / 256, 256, 0, stream>>>(Wk, WkT, 128, 128);
    wconv_t_kernel<<<(128 * 128 + 255) / 256, 256, 0, stream>>>(Wv, WvT, 128, 128);
    wconv_t_kernel<<<(128 * 128 + 255) / 256, 256, 0, stream>>>(proj_W, projT, 128, 128);
    wconv_t_kernel<<<(256 * 256 + 255) / 256, 256, 0, stream>>>(h1_W, h1T, 256, 256);
    wconv_t_kernel<<<(256 * 256 + 255) / 256, 256, 0, stream>>>(h2_W, h2T, 256, 256);

    // fused embed + attention: 2048 WGs x 256 threads, ~186 KB dynamic LDS
    const size_t lds1 = 10624 * sizeof(float) + (16 * 136 + 16 * 136 + 512 * 136) * sizeof(_Float16);
    hipFuncSetAttribute((const void*)attn_embed_kernel,
                        hipFuncAttributeMaxDynamicSharedMemorySize, (int)lds1);
    attn_embed_kernel<<<B_TOTAL / 16, 256, lds1, stream>>>(
        obs, own_W, own_b, int_W, int_b, v_att, proj_b, WqT, WkT, WvT, projT, x);

    // MLP head: 512 WGs x 256 threads, ~66 KB dynamic LDS
    const size_t lds2 = 2 * 64 * 264 * sizeof(_Float16);
    hipFuncSetAttribute((const void*)mlp_kernel,
                        hipFuncAttributeMaxDynamicSharedMemorySize, (int)lds2);
    mlp_kernel<<<B_TOTAL / 64, 256, lds2, stream>>>(
        x, h1T, h2T, h1_b, h2_b, out_W, out_b, out);
}
